// MultiheadAttention_32418413150910
// MI455X (gfx1250) — matrix operand order repaired and checked by reading, not compiled, not run on hardware
//
#include <hip/hip_runtime.h>
#include <hip/hip_bf16.h>

typedef __attribute__((ext_vector_type(16))) _Float16 v16h;
typedef __attribute__((ext_vector_type(8)))  float    v8f;

#define SEQ_N  2048
#define DMODEL 1024
#define NHEAD  16
#define DK     64
#define BATCH  2

#define WMMA_F32_F16(a,b,c) \
  __builtin_amdgcn_wmma_f32_16x16x32_f16(false,(a),false,(b),(short)0,(c),false,false)

// ---- WMMA fragment loaders (16-bit, wave32 layouts per CDNA5 ISA 7.12.2) ----
// A 16x32: lane L holds row M=L%16; K = 8*(L>=16) + {0..7} and 16 + 8*(L>=16) + {0..7}
__device__ __forceinline__ v16h frag_a_ld(const _Float16* __restrict__ row, int lane) {
  const int kb = (lane >= 16) ? 8 : 0;
  v16h r;
  uint4* d = reinterpret_cast<uint4*>(&r);
  d[0] = *reinterpret_cast<const uint4*>(row + kb);
  d[1] = *reinterpret_cast<const uint4*>(row + 16 + kb);
  return r;
}
// B 32x16 stored as Bt[n][k]: lane L holds col N=L%16; K = 8*(L>=16) + {0..7} and 16 + 8*(L>=16) + {0..7}
__device__ __forceinline__ v16h frag_b_ld(const _Float16* __restrict__ row, int lane) {
  const int kb = (lane >= 16) ? 8 : 0;
  v16h r;
  uint4* d = reinterpret_cast<uint4*>(&r);
  d[0] = *reinterpret_cast<const uint4*>(row + kb);
  d[1] = *reinterpret_cast<const uint4*>(row + 16 + kb);
  return r;
}

// Async 16B copy global -> LDS (CDNA5 GLOBAL_LOAD_ASYNC_TO_LDS_B128, ASYNCcnt).
// LDS offset = low 32 bits of the flat shared address (ISA 10.2 aperture map).
__device__ __forceinline__ void async_ld_b128(void* lds, const void* g) {
  asm volatile("global_load_async_to_lds_b128 %0, %1, off"
               :: "v"((unsigned)(size_t)lds), "v"(g) : "memory");
}
__device__ __forceinline__ void wait_asynccnt0() {
  asm volatile("s_wait_asynccnt 0x0" ::: "memory");
}

// =====================================================================
// GEMM: out = A(f32)[M,K] @ W(f32)[N,K]^T + bias   (W row-major == Bt)
// mode 0: write f16 scattered to [B, H, N, dk]   (head split for q/k/v)
// mode 1: write f32 row-major [M, N]             (final output proj)
// tile 128x128, K-stage 64 (2 wmma K-steps per barrier pair),
// 8 waves (4x2), each wave 32x64 -> 16 wmma per stage
// =====================================================================
#define GTM 128
#define GTN 128
#define GTK 64
#define LDSP 72   // padded LDS row stride (halves); 144B keeps 16B alignment

__global__ __launch_bounds__(256) void gemm_bias_kernel(
    const float* __restrict__ A, const float* __restrict__ W,
    const float* __restrict__ bias, _Float16* __restrict__ out16,
    float* __restrict__ out32, int M, int N, int K, int mode)
{
  __shared__ _Float16 As[GTM][LDSP];
  __shared__ _Float16 Bs[GTN][LDSP];

  const int tid  = threadIdx.x;
  const int lane = tid & 31;
  const int wave = tid >> 5;
  const int wm = (wave & 3) * 32;   // wave row offset in tile
  const int wn = (wave >> 2) * 64;  // wave col offset in tile
  const int bn = blockIdx.x * GTN;
  const int bm = blockIdx.y * GTM;
  const int lcol16 = lane & 15;
  const int halfsel = (lane >= 16) ? 8 : 0;

  v8f acc[2][4];
  const v8f vz = {0.f,0.f,0.f,0.f,0.f,0.f,0.f,0.f};
#pragma unroll
  for (int s = 0; s < 2; s++)
#pragma unroll
    for (int t = 0; t < 4; t++) acc[s][t] = vz;

  const int lrow = tid >> 1;          // 0..127
  const int lcol = (tid & 1) * 32;    // 0 or 32

  const float* ag = A + (size_t)(bm + lrow) * K + lcol;
  const float* wg = W + (size_t)(bn + lrow) * K + lcol;

  for (int kt = 0; kt < K; kt += GTK) {
    __syncthreads();
    { // stage A tile, f32 -> f16
      float fa[32];
#pragma unroll
      for (int i = 0; i < 8; i++)
        *reinterpret_cast<float4*>(&fa[i*4]) =
            *reinterpret_cast<const float4*>(ag + kt + i*4);
      _Float16 ha[32];
#pragma unroll
      for (int i = 0; i < 32; i++) ha[i] = (_Float16)fa[i];
#pragma unroll
      for (int i = 0; i < 4; i++)
        *reinterpret_cast<uint4*>(&As[lrow][lcol + i*8]) =
            *reinterpret_cast<uint4*>(&ha[i*8]);
    }
    { // stage W tile (Bt rows), f32 -> f16
      float fb[32];
#pragma unroll
      for (int i = 0; i < 8; i++)
        *reinterpret_cast<float4*>(&fb[i*4]) =
            *reinterpret_cast<const float4*>(wg + kt + i*4);
      _Float16 hb[32];
#pragma unroll
      for (int i = 0; i < 32; i++) hb[i] = (_Float16)fb[i];
#pragma unroll
      for (int i = 0; i < 4; i++)
        *reinterpret_cast<uint4*>(&Bs[lrow][lcol + i*8]) =
            *reinterpret_cast<uint4*>(&hb[i*8]);
    }
    if (kt + GTK < K) { // prefetch next K-slab into caches
      __builtin_prefetch(ag + kt + GTK, 0, 0);
      __builtin_prefetch(wg + kt + GTK, 0, 0);
    }
    __syncthreads();

#pragma unroll
    for (int kk = 0; kk < 2; kk++) {
      v16h af[2], bf[4];
#pragma unroll
      for (int s = 0; s < 2; s++)
        af[s] = frag_a_ld(&As[wm + s*16 + lcol16][kk*32], lane);
#pragma unroll
      for (int t = 0; t < 4; t++)
        bf[t] = frag_b_ld(&Bs[wn + t*16 + lcol16][kk*32], lane);
#pragma unroll
      for (int s = 0; s < 2; s++)
#pragma unroll
        for (int t = 0; t < 4; t++)
          acc[s][t] = WMMA_F32_F16(af[s], bf[t], acc[s][t]);
    }
  }

  // epilogue: C-layout (VGPR r: row = r + 8*(lane>=16); col = lane%16)
#pragma unroll
  for (int s = 0; s < 2; s++) {
#pragma unroll
    for (int t = 0; t < 4; t++) {
#pragma unroll
      for (int r = 0; r < 8; r++) {
        const int mrow = bm + wm + s*16 + r + halfsel;
        const int ncol = bn + wn + t*16 + lcol16;
        const float val = acc[s][t][r] + bias[ncol];
        if (mode == 0) {
          const int bb = mrow >> 11;            // / SEQ_N
          const int nn = mrow & (SEQ_N - 1);
          const int hh = ncol >> 6;             // / DK
          const int dd = ncol & (DK - 1);
          out16[(((size_t)bb * NHEAD + hh) * SEQ_N + nn) * DK + dd] = (_Float16)val;
        } else {
          out32[(size_t)mrow * N + ncol] = val;
        }
      }
    }
  }
}

// =====================================================================
// Flash attention: per-(b,h) streaming softmax(Q K^T masked) V
// 1 WG = 128 q rows (8 waves x 16 rows); K-tiles of 64 columns
// K-tile + mask-tile staged via GLOBAL_LOAD_ASYNC_TO_LDS_B128 (ASYNCcnt)
// =====================================================================
#define FQT 128
#define FKT 64
#define FP  72    // padded LDS row stride (halves); 144B keeps 16B alignment

__global__ __launch_bounds__(256) void flash_attn_kernel(
    const _Float16* __restrict__ Qh, const _Float16* __restrict__ Kh,
    const _Float16* __restrict__ Vh, const unsigned char* __restrict__ mask,
    float* __restrict__ ctx)
{
  __shared__ _Float16 Ks[FKT][FP];        // K tile [n][d]  (== Bt for S=Q K^T)
  __shared__ _Float16 Vt[DK][FP];         // V tile transposed [d][n] (== Bt for O=P V)
  __shared__ _Float16 Ps[8][16][FP];      // per-wave P (C->A relayout buffer)
  __shared__ unsigned char Ms[FQT][FKT];  // mask tile

  const int tid  = threadIdx.x;
  const int lane = tid & 31;
  const int wave = tid >> 5;
  const int bh = blockIdx.y;              // 0..B*H-1
  const int b  = bh >> 4;
  const int h  = bh & (NHEAD - 1);
  const int q0 = blockIdx.x * FQT;
  const int halfsel = (lane >= 16) ? 8 : 0;
  const int lcol = lane & 15;

  const size_t headBase = (size_t)bh * SEQ_N * DK;

  // Q fragments for this wave's 16 rows (held in registers for all K-tiles)
  v16h qa[2];
  {
    const _Float16* qrow = Qh + headBase + (size_t)(q0 + wave*16 + lcol) * DK;
    qa[0] = frag_a_ld(qrow, lane);
    qa[1] = frag_a_ld(qrow + 32, lane);
  }

  v8f o[4];
  const v8f vz = {0.f,0.f,0.f,0.f,0.f,0.f,0.f,0.f};
#pragma unroll
  for (int u = 0; u < 4; u++) o[u] = vz;
  float mrun[8], lrun[8];
#pragma unroll
  for (int r = 0; r < 8; r++) { mrun[r] = -1e30f; lrun[r] = 0.f; }

  const int krow = tid >> 2;          // 0..63
  const int kc   = (tid & 3) * 16;    // 0,16,32,48
  const int mr   = tid >> 1;          // 0..127
  const int mc   = (tid & 1) * 32;    // 0,32

  for (int k0 = 0; k0 < SEQ_N; k0 += FKT) {
    __syncthreads();
    { // K tile: async DMA global -> LDS (no VGPR round trip)
      const _Float16* kg = Kh + headBase + (size_t)(k0 + krow) * DK + kc;
      async_ld_b128(&Ks[krow][kc],     kg);
      async_ld_b128(&Ks[krow][kc + 8], kg + 8);
    }
    { // mask tile: async DMA (broadcast over heads; hits L2)
      const unsigned char* mg = mask + (size_t)b * SEQ_N * SEQ_N
                              + (size_t)(q0 + mr) * SEQ_N + k0 + mc;
      async_ld_b128(&Ms[mr][mc],      mg);
      async_ld_b128(&Ms[mr][mc + 16], mg + 16);
    }
    { // V tile, transposed into LDS (transform -> must go through VGPRs)
      const _Float16* vg = Vh + headBase + (size_t)(k0 + krow) * DK + kc;
      _Float16 hv[16];
      *reinterpret_cast<uint4*>(&hv[0]) = *reinterpret_cast<const uint4*>(vg);
      *reinterpret_cast<uint4*>(&hv[8]) = *reinterpret_cast<const uint4*>(vg + 8);
#pragma unroll
      for (int i = 0; i < 16; i++) Vt[kc + i][krow] = hv[i];
    }
    wait_asynccnt0();
    __syncthreads();

    // S = Q K^T for four 16-col subtiles
    v8f sfr[4];
#pragma unroll
    for (int t = 0; t < 4; t++) {
      v8f s = vz;
      v16h b0 = frag_b_ld(&Ks[t*16 + lcol][0],  lane);
      v16h b1 = frag_b_ld(&Ks[t*16 + lcol][32], lane);
      s = WMMA_F32_F16(qa[0], b0, s);
      s = WMMA_F32_F16(qa[1], b1, s);
      sfr[t] = s;
    }

    // mask + online softmax (row stats via 16-lane xor reductions)
#pragma unroll
    for (int r = 0; r < 8; r++) {
      const int lr = wave*16 + r + halfsel;   // row within Ms
      float rmax = -1e30f;
#pragma unroll
      for (int t = 0; t < 4; t++) {
        float vsc = sfr[t][r];
        if (Ms[lr][t*16 + lcol]) vsc = -1e9f;
        sfr[t][r] = vsc;
        rmax = fmaxf(rmax, vsc);
      }
#pragma unroll
      for (int off = 1; off < 16; off <<= 1)
        rmax = fmaxf(rmax, __shfl_xor(rmax, off, 32));
      const float nm = fmaxf(mrun[r], rmax);
      const float sc = __expf(mrun[r] - nm);
      float rsum = 0.f;
#pragma unroll
      for (int t = 0; t < 4; t++) {
        const float p = __expf(sfr[t][r] - nm);
        sfr[t][r] = p;
        rsum += p;
      }
#pragma unroll
      for (int off = 1; off < 16; off <<= 1)
        rsum += __shfl_xor(rsum, off, 32);
      lrun[r] = lrun[r] * sc + rsum;
      mrun[r] = nm;
#pragma unroll
      for (int u = 0; u < 4; u++) o[u][r] *= sc;
    }

    // P: C-layout -> A-layout via per-wave LDS (same-wave LDS is in-order)
#pragma unroll
    for (int r = 0; r < 8; r++) {
      const int lr = r + halfsel;
#pragma unroll
      for (int t = 0; t < 4; t++)
        Ps[wave][lr][t*16 + lcol] = (_Float16)sfr[t][r];
    }

    // O += P @ V
#pragma unroll
    for (int ks = 0; ks < 2; ks++) {
      v16h pa = frag_a_ld(&Ps[wave][lcol][ks*32], lane);
#pragma unroll
      for (int u = 0; u < 4; u++) {
        v16h bv = frag_b_ld(&Vt[u*16 + lcol][ks*32], lane);
        o[u] = WMMA_F32_F16(pa, bv, o[u]);
      }
    }
  }

  // normalize and write ctx as [B, N, D] f32 (input to output projection)
#pragma unroll
  for (int r = 0; r < 8; r++) {
    const float inv = 1.0f / lrun[r];
    const int qrow = q0 + wave*16 + r + halfsel;
#pragma unroll
    for (int u = 0; u < 4; u++) {
      const int dcol = h * DK + u*16 + lcol;
      ctx[((size_t)b * SEQ_N + qrow) * DMODEL + dcol] = o[u][r] * inv;
    }
  }
}

// =====================================================================
extern "C" void kernel_launch(void* const* d_in, const int* in_sizes, int n_in,
                              void* d_out, int out_size, void* d_ws, size_t ws_size,
                              hipStream_t stream) {
  (void)in_sizes; (void)n_in; (void)out_size; (void)ws_size;
  const float* q  = (const float*)d_in[0];
  const float* k  = (const float*)d_in[1];
  const float* v  = (const float*)d_in[2];
  const unsigned char* mask = (const unsigned char*)d_in[3];
  const float* Wq = (const float*)d_in[4];
  const float* bq = (const float*)d_in[5];
  const float* Wk = (const float*)d_in[6];
  const float* bk = (const float*)d_in[7];
  const float* Wv = (const float*)d_in[8];
  const float* bv = (const float*)d_in[9];
  const float* Wo = (const float*)d_in[10];
  const float* bo = (const float*)d_in[11];

  const int M = BATCH * SEQ_N;  // 4096
  const size_t headElems = (size_t)BATCH * NHEAD * SEQ_N * DK; // 4M f16 elems

  _Float16* qh = (_Float16*)d_ws;
  _Float16* kh = qh + headElems;
  _Float16* vh = kh + headElems;
  float*   ctx = (float*)(vh + headElems);   // [B,N,D] f32

  dim3 ggrid(DMODEL / GTN, M / GTM);         // (8, 32)
  dim3 gblk(256);

  gemm_bias_kernel<<<ggrid, gblk, 0, stream>>>(q, Wq, bq, qh, nullptr,
                                               M, DMODEL, DMODEL, 0);
  gemm_bias_kernel<<<ggrid, gblk, 0, stream>>>(k, Wk, bk, kh, nullptr,
                                               M, DMODEL, DMODEL, 0);
  gemm_bias_kernel<<<ggrid, gblk, 0, stream>>>(v, Wv, bv, vh, nullptr,
                                               M, DMODEL, DMODEL, 0);

  dim3 fgrid(SEQ_N / FQT, BATCH * NHEAD);    // (16, 32)
  flash_attn_kernel<<<fgrid, gblk, 0, stream>>>(qh, kh, vh, mask, ctx);

  gemm_bias_kernel<<<ggrid, gblk, 0, stream>>>(ctx, Wo, bo, nullptr, (float*)d_out,
                                               M, DMODEL, DMODEL, 1);
}
